// MiniMaxM2SparseMoeBlock_5093831213102
// MI455X (gfx1250) — compile-verified, matrix operand order checked by reading, not run on hardware
//
#include <hip/hip_runtime.h>
#include <hip/hip_bf16.h>
#include <stdint.h>

#define T_TOK 512
#define H_DIM 2048
#define I_DIM 768
#define E_NUM 16
#define QBLK  128

typedef __bf16 bf16_t;
typedef __bf16 v16bf __attribute__((ext_vector_type(16)));
typedef float  v8f   __attribute__((ext_vector_type(8)));

union FragB {
  v16bf v;
  uint4 q[2];
};

// async copy 16B global -> LDS (gfx1250), tracked by ASYNCcnt.
// The LDS pointer is passed through the asm (ptrtoint) so the shared array
// escapes and the compiler keeps the subsequent ds_load reads alive.
// Low 32 bits of a generic LDS address are the LDS byte offset (ISA aperture).
__device__ __forceinline__ void async_cp16(void* lds_ptr, const void* g) {
  asm volatile("global_load_async_to_lds_b128 %0, %1, off"
               :: "v"((unsigned)(uintptr_t)lds_ptr),
                  "v"((unsigned long long)(uintptr_t)g)
               : "memory");
}
__device__ __forceinline__ void async_wait0() {
  asm volatile("s_wait_asynccnt 0x0" ::: "memory");
}

// ---------------------------------------------------------------------------
// Kernel 0: zero output, zero counts, convert activations fp32 -> bf16
// ---------------------------------------------------------------------------
__global__ void moe_init(const float* __restrict__ x, float* __restrict__ y,
                         bf16_t* __restrict__ xb, int* __restrict__ counts) {
  int idx = blockIdx.x * blockDim.x + threadIdx.x;
  if (idx < T_TOK * H_DIM) {
    y[idx]  = 0.0f;
    xb[idx] = (bf16_t)x[idx];
  }
  if (idx < E_NUM) counts[idx] = 0;
}

// ---------------------------------------------------------------------------
// Kernel 1: router — sigmoid scores, top-2, normalized weights, gather lists
// ---------------------------------------------------------------------------
__global__ __launch_bounds__(128)
void moe_router(const float* __restrict__ x, const float* __restrict__ gw,
                int* __restrict__ counts, int* __restrict__ tok_idx,
                float* __restrict__ tok_w) {
  int wave = threadIdx.x >> 5;
  int lane = threadIdx.x & 31;
  int t = blockIdx.x * 4 + wave;
  if (t >= T_TOK) return;

  float4 xr[16];
  const float4* xp = (const float4*)(x + (size_t)t * H_DIM);
#pragma unroll
  for (int i = 0; i < 16; ++i) xr[i] = xp[lane + i * 32];

  float best = -1e30f, second = -1e30f;
  int bi = 0, si = 0;
  for (int e = 0; e < E_NUM; ++e) {
    const float4* gp = (const float4*)(gw + (size_t)e * H_DIM);
    float acc = 0.f;
#pragma unroll
    for (int i = 0; i < 16; ++i) {
      float4 g = gp[lane + i * 32];
      acc += xr[i].x * g.x + xr[i].y * g.y + xr[i].z * g.z + xr[i].w * g.w;
    }
#pragma unroll
    for (int off = 16; off >= 1; off >>= 1) acc += __shfl_xor(acc, off, 32);
    float s = 1.0f / (1.0f + __expf(-acc));
    if (s > best)        { second = best; si = bi; best = s; bi = e; }
    else if (s > second) { second = s; si = e; }
  }
  if (lane == 0) {
    float wsum = best + second;
    float w0 = best / wsum, w1 = second / wsum;
    int p0 = atomicAdd(&counts[bi], 1);
    tok_idx[bi * T_TOK + p0] = t;
    tok_w[bi * T_TOK + p0]   = w0;
    int p1 = atomicAdd(&counts[si], 1);
    tok_idx[si * T_TOK + p1] = t;
    tok_w[si * T_TOK + p1]   = w1;
  }
}

// ---------------------------------------------------------------------------
// Kernel 2: tiny exclusive scan over 16 expert counts
// ---------------------------------------------------------------------------
__global__ void moe_scan(const int* __restrict__ counts, int* __restrict__ offs) {
  if (threadIdx.x == 0) {
    int r = 0;
    for (int e = 0; e < E_NUM; ++e) { offs[e] = r; r += counts[e]; }
  }
}

// ---------------------------------------------------------------------------
// Kernel 3: gathered GEMM1 — act = silu(x@W1^T) * (x@W3^T), bf16 out
// grid (I/128, T/64, E); block 256 = 8 waves; wave tile = 16 I x 64 tokens
// ---------------------------------------------------------------------------
__global__ __launch_bounds__(256)
void moe_gemm1(const bf16_t* __restrict__ xb,
               const float* __restrict__ w1, const float* __restrict__ w1s,
               const float* __restrict__ w3, const float* __restrict__ w3s,
               const int* __restrict__ counts, const int* __restrict__ offs,
               const int* __restrict__ tok_idx, bf16_t* __restrict__ act) {
  int e     = blockIdx.z;
  int tileN = blockIdx.y;
  int cnt   = counts[e];
  if (tileN * 64 >= cnt) return;

  int band = blockIdx.x * 128;
  int wave = threadIdx.x >> 5;
  int lane = threadIdx.x & 31;
  int row  = band + wave * 16 + (lane & 15);  // A row (I dim)
  int hi   = lane >> 4;                       // kbase = hi*8 (ISA A/B layout)

  __shared__ bf16_t xs[64][136];              // 64 tok x 128 K bf16, padded

  const float* w1row = w1 + ((size_t)e * I_DIM + row) * H_DIM;
  const float* w3row = w3 + ((size_t)e * I_DIM + row) * H_DIM;
  const float* s1row = w1s + ((size_t)e * I_DIM + row) * (H_DIM / QBLK);
  const float* s3row = w3s + ((size_t)e * I_DIM + row) * (H_DIM / QBLK);

  v8f gc[4] = {{}, {}, {}, {}};
  v8f uc[4] = {{}, {}, {}, {}};

  int ltok  = threadIdx.x >> 2;               // 0..63 token for staging
  int lpart = threadIdx.x & 3;                // 0..3, 32 halves each
  int spos  = tileN * 64 + ltok;
  int tglob = (spos < cnt) ? tok_idx[e * T_TOK + spos] : 0;
  const bf16_t* xrow = xb + (size_t)tglob * H_DIM;
  bf16_t* ldsdst = &xs[ltok][lpart * 32];

  for (int kb = 0; kb < H_DIM / QBLK; ++kb) {
    // async-stage gathered X tile (bf16) into LDS: 64B per thread
    {
      const char* src = (const char*)(xrow + kb * QBLK + lpart * 32);
#pragma unroll
      for (int i = 0; i < 4; ++i) async_cp16(ldsdst + i * 8, src + i * 16);
      async_wait0();
    }
    __syncthreads();

    float sc1 = s1row[kb];
    float sc3 = s3row[kb];
    if (kb + 1 < H_DIM / QBLK) {
      __builtin_prefetch(w1row + (size_t)(kb + 1) * QBLK, 0, 0);
      __builtin_prefetch(w3row + (size_t)(kb + 1) * QBLK, 0, 0);
    }

#pragma unroll
    for (int kc = 0; kc < 4; ++kc) {
      int kA = kb * QBLK + kc * 32 + hi * 8;
      v16bf a1, a3;
      {
        const float4* p = (const float4*)(w1row + kA);
        float4 f0 = p[0], f1 = p[1], f2 = p[4], f3 = p[5];
        a1[0] = (bf16_t)(f0.x * sc1); a1[1] = (bf16_t)(f0.y * sc1);
        a1[2] = (bf16_t)(f0.z * sc1); a1[3] = (bf16_t)(f0.w * sc1);
        a1[4] = (bf16_t)(f1.x * sc1); a1[5] = (bf16_t)(f1.y * sc1);
        a1[6] = (bf16_t)(f1.z * sc1); a1[7] = (bf16_t)(f1.w * sc1);
        a1[8]  = (bf16_t)(f2.x * sc1); a1[9]  = (bf16_t)(f2.y * sc1);
        a1[10] = (bf16_t)(f2.z * sc1); a1[11] = (bf16_t)(f2.w * sc1);
        a1[12] = (bf16_t)(f3.x * sc1); a1[13] = (bf16_t)(f3.y * sc1);
        a1[14] = (bf16_t)(f3.z * sc1); a1[15] = (bf16_t)(f3.w * sc1);
      }
      {
        const float4* p = (const float4*)(w3row + kA);
        float4 f0 = p[0], f1 = p[1], f2 = p[4], f3 = p[5];
        a3[0] = (bf16_t)(f0.x * sc3); a3[1] = (bf16_t)(f0.y * sc3);
        a3[2] = (bf16_t)(f0.z * sc3); a3[3] = (bf16_t)(f0.w * sc3);
        a3[4] = (bf16_t)(f1.x * sc3); a3[5] = (bf16_t)(f1.y * sc3);
        a3[6] = (bf16_t)(f1.z * sc3); a3[7] = (bf16_t)(f1.w * sc3);
        a3[8]  = (bf16_t)(f2.x * sc3); a3[9]  = (bf16_t)(f2.y * sc3);
        a3[10] = (bf16_t)(f2.z * sc3); a3[11] = (bf16_t)(f2.w * sc3);
        a3[12] = (bf16_t)(f3.x * sc3); a3[13] = (bf16_t)(f3.y * sc3);
        a3[14] = (bf16_t)(f3.z * sc3); a3[15] = (bf16_t)(f3.w * sc3);
      }
#pragma unroll
      for (int sub = 0; sub < 4; ++sub) {
        FragB b;
        b.q[0] = *(const uint4*)&xs[(lane & 15) + sub * 16][kc * 32 + hi * 8];
        b.q[1] = *(const uint4*)&xs[(lane & 15) + sub * 16][kc * 32 + hi * 8 + 16];
        gc[sub] = __builtin_amdgcn_wmma_f32_16x16x32_bf16(
            false, a1, false, b.v, (short)0, gc[sub], false, false);
        uc[sub] = __builtin_amdgcn_wmma_f32_16x16x32_bf16(
            false, a3, false, b.v, (short)0, uc[sub], false, false);
      }
    }
    __syncthreads();
  }

  // Epilogue: SwiGLU, pack bf16, one 16B store per C-fragment per lane
  int i0 = band + wave * 16 + hi * 8;
  int n  = lane & 15;
#pragma unroll
  for (int sub = 0; sub < 4; ++sub) {
    int pos = tileN * 64 + sub * 16 + n;
    if (pos < cnt) {
      int slot = offs[e] + pos;
      union { uint4 q; bf16_t h[8]; } o;
#pragma unroll
      for (int r = 0; r < 8; ++r) {
        float g = gc[sub][r], u = uc[sub][r];
        o.h[r] = (bf16_t)(g / (1.0f + __expf(-g)) * u);
      }
      *(uint4*)&act[(size_t)slot * I_DIM + i0] = o.q;
    }
  }
}

// ---------------------------------------------------------------------------
// Kernel 4: gathered GEMM2 — y[t] += tw * (act @ W2^T)
// grid (H/128, T/64, E); block 256 = 8 waves
// ---------------------------------------------------------------------------
__global__ __launch_bounds__(256)
void moe_gemm2(const bf16_t* __restrict__ act,
               const float* __restrict__ w2, const float* __restrict__ w2s,
               const int* __restrict__ counts, const int* __restrict__ offs,
               const int* __restrict__ tok_idx, const float* __restrict__ tok_w,
               float* __restrict__ y) {
  int e     = blockIdx.z;
  int tileN = blockIdx.y;
  int cnt   = counts[e];
  if (tileN * 64 >= cnt) return;

  int band = blockIdx.x * 128;
  int wave = threadIdx.x >> 5;
  int lane = threadIdx.x & 31;
  int row  = band + wave * 16 + (lane & 15);  // H row
  int hi   = lane >> 4;

  __shared__ bf16_t as_[64][136];

  const float* wrow = w2 + ((size_t)e * H_DIM + row) * I_DIM;
  const float* srow = w2s + ((size_t)e * H_DIM + row) * (I_DIM / QBLK);

  v8f c[4] = {{}, {}, {}, {}};

  int ltok  = threadIdx.x >> 2;
  int lpart = threadIdx.x & 3;
  int spos  = tileN * 64 + ltok;
  int slot  = offs[e] + ((spos < cnt) ? spos : 0);
  const bf16_t* arow = act + (size_t)slot * I_DIM;
  bf16_t* ldsdst = &as_[ltok][lpart * 32];

  for (int kb = 0; kb < I_DIM / QBLK; ++kb) {   // 6 blocks of 128
    {
      const char* src = (const char*)(arow + kb * QBLK + lpart * 32);
#pragma unroll
      for (int i = 0; i < 4; ++i) async_cp16(ldsdst + i * 8, src + i * 16);
      async_wait0();
    }
    __syncthreads();

    float sc = srow[kb];
    if (kb + 1 < I_DIM / QBLK)
      __builtin_prefetch(wrow + (size_t)(kb + 1) * QBLK, 0, 0);

#pragma unroll
    for (int kc = 0; kc < 4; ++kc) {
      int kA = kb * QBLK + kc * 32 + hi * 8;
      v16bf a;
      {
        const float4* p = (const float4*)(wrow + kA);
        float4 f0 = p[0], f1 = p[1], f2 = p[4], f3 = p[5];
        a[0] = (bf16_t)(f0.x * sc); a[1] = (bf16_t)(f0.y * sc);
        a[2] = (bf16_t)(f0.z * sc); a[3] = (bf16_t)(f0.w * sc);
        a[4] = (bf16_t)(f1.x * sc); a[5] = (bf16_t)(f1.y * sc);
        a[6] = (bf16_t)(f1.z * sc); a[7] = (bf16_t)(f1.w * sc);
        a[8]  = (bf16_t)(f2.x * sc); a[9]  = (bf16_t)(f2.y * sc);
        a[10] = (bf16_t)(f2.z * sc); a[11] = (bf16_t)(f2.w * sc);
        a[12] = (bf16_t)(f3.x * sc); a[13] = (bf16_t)(f3.y * sc);
        a[14] = (bf16_t)(f3.z * sc); a[15] = (bf16_t)(f3.w * sc);
      }
#pragma unroll
      for (int sub = 0; sub < 4; ++sub) {
        FragB b;
        b.q[0] = *(const uint4*)&as_[(lane & 15) + sub * 16][kc * 32 + hi * 8];
        b.q[1] = *(const uint4*)&as_[(lane & 15) + sub * 16][kc * 32 + hi * 8 + 16];
        c[sub] = __builtin_amdgcn_wmma_f32_16x16x32_bf16(
            false, a, false, b.v, (short)0, c[sub], false, false);
      }
    }
    __syncthreads();
  }

  // Epilogue: scale by router weight, atomic scatter into y (<=2 adds/elem)
  int h0 = band + wave * 16 + hi * 8;
  int n  = lane & 15;
#pragma unroll
  for (int sub = 0; sub < 4; ++sub) {
    int pos = tileN * 64 + sub * 16 + n;
    if (pos < cnt) {
      int tk = tok_idx[e * T_TOK + pos];
      float tw = tok_w[e * T_TOK + pos];
#pragma unroll
      for (int r = 0; r < 8; ++r)
        atomicAdd(&y[(size_t)tk * H_DIM + h0 + r], c[sub][r] * tw);
    }
  }
}

// ---------------------------------------------------------------------------
extern "C" void kernel_launch(void* const* d_in, const int* in_sizes, int n_in,
                              void* d_out, int out_size, void* d_ws, size_t ws_size,
                              hipStream_t stream) {
  const float* x   = (const float*)d_in[0];   // [1,512,2048]
  const float* gw  = (const float*)d_in[1];   // [16,2048]
  const float* w1  = (const float*)d_in[2];   // [16,768,2048]
  const float* w1s = (const float*)d_in[3];   // [16,768,16]
  const float* w3  = (const float*)d_in[4];
  const float* w3s = (const float*)d_in[5];
  const float* w2  = (const float*)d_in[6];   // [16,2048,768]
  const float* w2s = (const float*)d_in[7];   // [16,2048,6]
  float* y = (float*)d_out;                   // [1,512,2048]

  char* ws = (char*)d_ws;
  int*    counts  = (int*)ws;                                    // pad 256
  int*    offs    = (int*)(ws + 256);                            // pad 256
  int*    tok_idx = (int*)(ws + 512);                            // 32 KB
  float*  tok_w   = (float*)(ws + 512 + (size_t)E_NUM * T_TOK * 4);
  bf16_t* xb      = (bf16_t*)(ws + 512 + 2 * (size_t)E_NUM * T_TOK * 4);
  bf16_t* actb    = (bf16_t*)((char*)xb + (size_t)T_TOK * H_DIM * 2);
  // total ~3.7 MB

  moe_init<<<(T_TOK * H_DIM + 255) / 256, 256, 0, stream>>>(x, y, xb, counts);
  moe_router<<<T_TOK / 4, 128, 0, stream>>>(x, gw, counts, tok_idx, tok_w);
  moe_scan<<<1, 32, 0, stream>>>(counts, offs);
  moe_gemm1<<<dim3(I_DIM / 128, T_TOK / 64, E_NUM), 256, 0, stream>>>(
      xb, w1, w1s, w3, w3s, counts, offs, tok_idx, actb);
  moe_gemm2<<<dim3(H_DIM / 128, T_TOK / 64, E_NUM), 256, 0, stream>>>(
      actb, w2, w2s, counts, offs, tok_idx, tok_w, y);
}
